// LocalContextAttention_8993661518093
// MI455X (gfx1250) — matrix operand order repaired and checked by reading, not compiled, not run on hardware
//
#include <hip/hip_runtime.h>
#include <hip/hip_bf16.h>

// ---------------------------------------------------------------------------
// Problem constants (reference: B=8, S=2048, D=768, H=8, HD=96)
// ---------------------------------------------------------------------------
constexpr int Bn = 8;
constexpr int Sn = 2048;
constexpr int Dn = 768;
constexpr int Hn = 8;
constexpr int HDn = 96;          // head dim
constexpr float LN_EPS = 1e-5f;

typedef __attribute__((ext_vector_type(16))) __bf16  v16bf;
typedef __attribute__((ext_vector_type(8)))  float   v8f;
typedef __attribute__((ext_vector_type(4)))  unsigned int u32x4;

#define WMMA_BF16(A, Bm, C) \
  __builtin_amdgcn_wmma_f32_16x16x32_bf16(false, (A), false, (Bm), (short)0, (C), false, false)

// Assemble a 16-element bf16 fragment from two 16-byte chunks (global or LDS).
static __device__ __forceinline__ v16bf load_frag(const __bf16* p0, const __bf16* p1) {
  v16bf r;
  ((u32x4*)&r)[0] = *(const u32x4*)p0;
  ((u32x4*)&r)[1] = *(const u32x4*)p1;
  return r;
}

// Async global -> LDS copy of 16 bytes (CDNA5, tracked by ASYNCcnt).
// lds_off is the byte offset within this block's LDS allocation (HW adds LDS_BASE).
static __device__ __forceinline__ void async_g2l_b128(unsigned lds_off, const void* gptr) {
  unsigned long long ga = (unsigned long long)gptr;
  asm volatile("global_load_async_to_lds_b128 %0, %1, off"
               :: "v"(lds_off), "v"(ga)
               : "memory");
}
static __device__ __forceinline__ void wait_asynccnt0() {
  asm volatile("s_wait_asynccnt 0x0" ::: "memory");
}

// ---------------------------------------------------------------------------
// Kernel 0: fp32 -> bf16 conversion
// ---------------------------------------------------------------------------
__global__ void cvt_bf16_kernel(const float* __restrict__ in, __bf16* __restrict__ out, int n) {
  int i = blockIdx.x * blockDim.x + threadIdx.x;
  if (i < n) out[i] = (__bf16)in[i];
}

// ---------------------------------------------------------------------------
// Kernel 1: QKV projection GEMM.  Y = X @ W^T + b, X:[16384,768] W:[768,768]
// One wave computes a 64x64 output block: 4 A-frags x 4 B-frags, 16 WMMAs per
// 32-wide K step (24 steps).  32 FLOP/byte of fetched data.
// Q,K stored [BH][S][96] bf16; V stored transposed [BH][96][S] bf16.
// ---------------------------------------------------------------------------
__global__ __launch_bounds__(128) void qkv_gemm_kernel(
    const __bf16* __restrict__ X,     // [B*S, D] bf16
    const __bf16* __restrict__ W,     // [3][D][D] bf16 (row n = output feature)
    const float*  __restrict__ bq,
    const float*  __restrict__ bk,
    const float*  __restrict__ bv,
    __bf16* __restrict__ Qb,          // [BH][S][96]
    __bf16* __restrict__ Kb,          // [BH][S][96]
    __bf16* __restrict__ Vt)          // [BH][96][S]
{
  constexpr int GM = (Bn * Sn) / 64;  // 256  (64-row groups)
  constexpr int GN = Dn / 64;         // 12   (64-col groups)

  const int lane = threadIdx.x & 31;
  const int wave = (blockIdx.x * blockDim.x + threadIdx.x) >> 5;
  const int matId = wave / (GM * GN);
  const int rem   = wave % (GM * GN);
  const int tm4 = rem / GN;
  const int tn4 = rem % GN;
  const int ln = lane & 15;
  const int hf = lane >> 4;

  const __bf16* Wm   = W + (size_t)matId * Dn * Dn;
  const float*  bias = (matId == 0) ? bq : ((matId == 1) ? bk : bv);

  const __bf16* Arow[4];
  const __bf16* Brow[4];
#pragma unroll
  for (int i = 0; i < 4; ++i)
    Arow[i] = X + (size_t)(tm4 * 64 + i * 16 + ln) * Dn;
#pragma unroll
  for (int j = 0; j < 4; ++j)
    Brow[j] = Wm + (size_t)(tn4 * 64 + j * 16 + ln) * Dn;

  v8f acc[4][4];
#pragma unroll
  for (int i = 0; i < 4; ++i)
#pragma unroll
    for (int j = 0; j < 4; ++j)
      acc[i][j] = (v8f){0.f,0.f,0.f,0.f,0.f,0.f,0.f,0.f};

  for (int kb = 0; kb < Dn; kb += 32) {
    v16bf a[4], b[4];
#pragma unroll
    for (int i = 0; i < 4; ++i)
      a[i] = load_frag(Arow[i] + kb + 16 * hf, Arow[i] + kb + 16 * hf + 8);
#pragma unroll
    for (int j = 0; j < 4; ++j)
      b[j] = load_frag(Brow[j] + kb + 16 * hf, Brow[j] + kb + 16 * hf + 8);
#pragma unroll
    for (int i = 0; i < 4; ++i)
#pragma unroll
      for (int j = 0; j < 4; ++j)
        acc[i][j] = WMMA_BF16(a[i], b[j], acc[i][j]);
  }

#pragma unroll
  for (int j = 0; j < 4; ++j) {
    const int ncol = tn4 * 64 + j * 16 + ln;
    const float bn = bias[ncol];
    const int h = ncol / HDn;
    const int d = ncol % HDn;
#pragma unroll
    for (int i = 0; i < 4; ++i) {
#pragma unroll
      for (int r = 0; r < 8; ++r) {
        const int m  = tm4 * 64 + i * 16 + r + 8 * hf;   // C layout: row = r + 8*half
        const int bi = m >> 11;                           // m / S
        const int s  = m & (Sn - 1);
        const int bh = bi * Hn + h;
        const __bf16 v = (__bf16)(acc[i][j][r] + bn);
        if (matId == 0)      Qb[((size_t)bh * Sn + s) * HDn + d] = v;
        else if (matId == 1) Kb[((size_t)bh * Sn + s) * HDn + d] = v;
        else                 Vt[((size_t)bh * HDn + d) * Sn + s] = v;
      }
    }
  }
}

// ---------------------------------------------------------------------------
// Kernel 2: flash-style masked attention.
// Block = 4 waves sharing one (b,h); each wave owns one 16-query tile
// (block covers 64 queries).  Per 32-key step the block stages the K tile
// (32x96 bf16) and V tile (96x32 bf16) into LDS with async global->LDS
// copies; each wave then does 6 score WMMAs, fp32 online softmax, an LDS
// transpose of P (C-layout -> A-layout), and 6 context WMMAs.
// ---------------------------------------------------------------------------
#define ATT_WPB 4
// LDS map (single static allocation; offsets are LDS byte addresses)
constexpr unsigned KL_OFF = 0;                    // 32*96*2   = 6144 B
constexpr unsigned VL_OFF = 6144;                 // 96*32*2   = 6144 B
constexpr unsigned P_OFF  = 12288;                // 4 * 16*32*2 = 4096 B
constexpr unsigned LDS_BYTES = 16384;

__global__ __launch_bounds__(32 * ATT_WPB) void attention_kernel(
    const __bf16* __restrict__ Qb,   // [BH][S][96]
    const __bf16* __restrict__ Kb,   // [BH][S][96]
    const __bf16* __restrict__ Vt,   // [BH][96][S]
    const int*    __restrict__ adj,  // [B][S][S]
    float*        __restrict__ Ctx)  // [B][S][D] fp32
{
  __shared__ __align__(16) unsigned char lds_raw[LDS_BYTES];

  const int tid   = threadIdx.x;
  const int lane  = tid & 31;
  const int wslot = tid >> 5;
  const int bh = blockIdx.x >> 5;            // 32 query-groups of 64 per (b,h)
  const int qg = blockIdx.x & 31;
  const int b  = bh >> 3;
  const int h  = bh & (Hn - 1);
  const int ln = lane & 15;
  const int hf = lane >> 4;
  const int qbase = qg * 64 + wslot * 16;

  const __bf16* Kl  = (const __bf16*)(lds_raw + KL_OFF);               // [32][96]
  const __bf16* Vl  = (const __bf16*)(lds_raw + VL_OFF);               // [96][32]
  __bf16*       pls = (__bf16*)(lds_raw + P_OFF) + wslot * 16 * 32;    // per-wave P

  // Q fragments: A-matrix 16x96 as three 16x32 fragments (loaded once).
  v16bf qa[3];
  {
    const __bf16* qrow = Qb + ((size_t)bh * Sn + qbase + ln) * HDn;
#pragma unroll
    for (int f = 0; f < 3; ++f)
      qa[f] = load_frag(qrow + f * 32 + 16 * hf, qrow + f * 32 + 16 * hf + 8);
  }

  v8f ctx[6];
#pragma unroll
  for (int dt = 0; dt < 6; ++dt) ctx[dt] = (v8f){0.f,0.f,0.f,0.f,0.f,0.f,0.f,0.f};
  float mrun[8], lrun[8];
#pragma unroll
  for (int r = 0; r < 8; ++r) { mrun[r] = -3.0e38f; lrun[r] = 0.f; }

  const float scale = 0.10206207261596575f;   // 1/sqrt(96)
  const int* adjb = adj + (size_t)b * Sn * Sn;
  const unsigned char* Kg = (const unsigned char*)Kb + (size_t)bh * Sn * HDn * 2;
  const unsigned char* Vg = (const unsigned char*)Vt + (size_t)bh * HDn * Sn * 2;

  for (int kb = 0; kb < Sn; kb += 32) {
    // ---- stage K tile [32][96] and V tile [96][32] into LDS (async) ----
    __syncthreads();   // previous step's LDS reads complete before overwrite
#pragma unroll
    for (int c = 0; c < 3; ++c) {
      // K tile: 32 rows * 192 B = 384 chunks of 16 B
      const int ck = tid + c * 128;
      const int kk = ck / 12, jk = ck % 12;
      async_g2l_b128(KL_OFF + (unsigned)(kk * 192 + jk * 16),
                     Kg + (size_t)(kb + kk) * 192 + jk * 16);
      // V tile: 96 rows * 64 B = 384 chunks of 16 B
      const int cv = tid + c * 128;
      const int dv = cv >> 2, jv = cv & 3;
      async_g2l_b128(VL_OFF + (unsigned)(dv * 64 + jv * 16),
                     Vg + (size_t)dv * (Sn * 2) + (size_t)kb * 2 + jv * 16);
    }
    wait_asynccnt0();
    __syncthreads();

    // ---- scores: two 16x16 tiles (keys kb..kb+15 and kb+16..kb+31) ----
    v8f s0 = {0.f,0.f,0.f,0.f,0.f,0.f,0.f,0.f};
    v8f s1 = {0.f,0.f,0.f,0.f,0.f,0.f,0.f,0.f};
#pragma unroll
    for (int f = 0; f < 3; ++f) {
      const __bf16* kp0 = Kl + ln * HDn        + f * 32 + 16 * hf;
      const __bf16* kp1 = Kl + (16 + ln) * HDn + f * 32 + 16 * hf;
      v16bf b0 = load_frag(kp0, kp0 + 8);
      v16bf b1 = load_frag(kp1, kp1 + 8);
      s0 = WMMA_BF16(qa[f], b0, s0);
      s1 = WMMA_BF16(qa[f], b1, s1);
    }

    // ---- mask + scale + online softmax (fp32) ----
    float p0[8], p1[8];
#pragma unroll
    for (int r = 0; r < 8; ++r) {
      const int q = qbase + r + 8 * hf;                   // C layout row
      const int* ar = adjb + (size_t)q * Sn + kb;
      float v0 = (ar[ln]      != 0) ? s0[r] * scale : -1e9f;
      float v1 = (ar[16 + ln] != 0) ? s1[r] * scale : -1e9f;
      float tmax = fmaxf(v0, v1);
      tmax = fmaxf(tmax, __shfl_xor(tmax, 1, 32));
      tmax = fmaxf(tmax, __shfl_xor(tmax, 2, 32));
      tmax = fmaxf(tmax, __shfl_xor(tmax, 4, 32));
      tmax = fmaxf(tmax, __shfl_xor(tmax, 8, 32));
      const float nm   = fmaxf(mrun[r], tmax);
      const float corr = __expf(mrun[r] - nm);
      mrun[r] = nm;
      p0[r] = __expf(v0 - nm);
      p1[r] = __expf(v1 - nm);
      float ts = p0[r] + p1[r];
      ts += __shfl_xor(ts, 1, 32);
      ts += __shfl_xor(ts, 2, 32);
      ts += __shfl_xor(ts, 4, 32);
      ts += __shfl_xor(ts, 8, 32);
      lrun[r] = lrun[r] * corr + ts;
#pragma unroll
      for (int dt = 0; dt < 6; ++dt) ctx[dt][r] *= corr;
    }

    // ---- transpose P (C layout) -> A layout via per-wave LDS slot ----
#pragma unroll
    for (int r = 0; r < 8; ++r) {
      const int row = r + 8 * hf;
      pls[row * 32 + ln]      = (__bf16)p0[r];
      pls[row * 32 + 16 + ln] = (__bf16)p1[r];
    }
    // Same-wave DS ops are in-order: safe to read back immediately.
    v16bf pa = load_frag(pls + ln * 32 + 16 * hf, pls + ln * 32 + 16 * hf + 8);

    // ---- context accumulate: ctx[16x96] += P[16x32] @ V[32x96] ----
#pragma unroll
    for (int dt = 0; dt < 6; ++dt) {
      const __bf16* vp = Vl + (dt * 16 + ln) * 32 + 16 * hf;
      v16bf vb = load_frag(vp, vp + 8);
      ctx[dt] = WMMA_BF16(pa, vb, ctx[dt]);
    }
  }

  // ---- epilogue: normalize and scatter into [B,S,D] fp32 ----
#pragma unroll
  for (int r = 0; r < 8; ++r) {
    const float inv = 1.0f / lrun[r];
    const int q = qbase + r + 8 * hf;
    float* crow = Ctx + ((size_t)(b * Sn + q)) * Dn + h * HDn;
#pragma unroll
    for (int dt = 0; dt < 6; ++dt)
      crow[dt * 16 + ln] = ctx[dt][r] * inv;
  }
}

// ---------------------------------------------------------------------------
// Kernel 3: fused residual add + LayerNorm.  One 256-thread block per row.
// ---------------------------------------------------------------------------
__global__ __launch_bounds__(256) void residual_ln_kernel(
    const float* __restrict__ Ctx,
    const float* __restrict__ feat,
    const float* __restrict__ gamma,
    const float* __restrict__ beta,
    float*       __restrict__ out)
{
  __shared__ float red_s[8];
  __shared__ float red_q[8];

  const size_t row = blockIdx.x;
  const float* c = Ctx  + row * Dn;
  const float* f = feat + row * Dn;

  float x[3];
  float sum = 0.f, sq = 0.f;
#pragma unroll
  for (int i = 0; i < 3; ++i) {
    const int d = threadIdx.x + i * 256;
    x[i] = c[d] + f[d];
    sum += x[i];
    sq  += x[i] * x[i];
  }
#pragma unroll
  for (int off = 1; off < 32; off <<= 1) {
    sum += __shfl_xor(sum, off, 32);
    sq  += __shfl_xor(sq,  off, 32);
  }
  const int w = threadIdx.x >> 5;
  if ((threadIdx.x & 31) == 0) { red_s[w] = sum; red_q[w] = sq; }
  __syncthreads();
  float tsum = 0.f, tsq = 0.f;
#pragma unroll
  for (int i = 0; i < 8; ++i) { tsum += red_s[i]; tsq += red_q[i]; }

  const float mu   = tsum * (1.0f / Dn);
  const float var  = tsq * (1.0f / Dn) - mu * mu;
  const float rstd = rsqrtf(var + LN_EPS);
#pragma unroll
  for (int i = 0; i < 3; ++i) {
    const int d = threadIdx.x + i * 256;
    out[row * Dn + d] = (x[i] - mu) * rstd * gamma[d] + beta[d];
  }
}

// ---------------------------------------------------------------------------
// Launch
// ---------------------------------------------------------------------------
extern "C" void kernel_launch(void* const* d_in, const int* in_sizes, int n_in,
                              void* d_out, int out_size, void* d_ws, size_t ws_size,
                              hipStream_t stream) {
  (void)in_sizes; (void)n_in; (void)out_size; (void)ws_size;

  const float* feat  = (const float*)d_in[0];
  const int*   adj   = (const int*)  d_in[1];
  const float* Wq    = (const float*)d_in[2];
  const float* bq    = (const float*)d_in[3];
  const float* Wk    = (const float*)d_in[4];
  const float* bk    = (const float*)d_in[5];
  const float* Wv    = (const float*)d_in[6];
  const float* bv    = (const float*)d_in[7];
  const float* gamma = (const float*)d_in[8];
  const float* beta  = (const float*)d_in[9];
  float* out = (float*)d_out;

  // Workspace layout
  constexpr size_t N_X = (size_t)Bn * Sn * Dn;   // 12,582,912
  constexpr size_t N_W = (size_t)Dn * Dn;        //    589,824
  char* ws = (char*)d_ws;
  __bf16* Xbf = (__bf16*)(ws);                                  // 25,165,824 B
  __bf16* Wbf = (__bf16*)(ws + N_X * 2);                        //  3,538,944 B
  __bf16* Qb  = (__bf16*)(ws + N_X * 2 + 3 * N_W * 2);          // 25,165,824 B
  __bf16* Kb  = Qb + N_X;                                       // 25,165,824 B
  __bf16* Vt  = Kb + N_X;                                       // 25,165,824 B
  float*  Ctx = (float*)(Vt + N_X);                             // 50,331,648 B

  // 0) fp32 -> bf16 conversions
  cvt_bf16_kernel<<<(int)((N_X + 255) / 256), 256, 0, stream>>>(feat, Xbf, (int)N_X);
  cvt_bf16_kernel<<<(int)((N_W + 255) / 256), 256, 0, stream>>>(Wq, Wbf + 0 * N_W, (int)N_W);
  cvt_bf16_kernel<<<(int)((N_W + 255) / 256), 256, 0, stream>>>(Wk, Wbf + 1 * N_W, (int)N_W);
  cvt_bf16_kernel<<<(int)((N_W + 255) / 256), 256, 0, stream>>>(Wv, Wbf + 2 * N_W, (int)N_W);

  // 1) QKV projections: 3 * 256 * 12 = 9216 waves, 4 waves/block
  {
    const int total_waves = 3 * ((Bn * Sn) / 64) * (Dn / 64);
    qkv_gemm_kernel<<<total_waves / 4, 128, 0, stream>>>(
        Xbf, Wbf, bq, bk, bv, Qb, Kb, Vt);
  }

  // 2) attention: one block per (b,h,64-query group) = 64 * 32 = 2048 blocks
  attention_kernel<<<Bn * Hn * (Sn / 64), 32 * ATT_WPB, 0, stream>>>(
      Qb, Kb, Vt, adj, Ctx);

  // 3) residual + LayerNorm: one block per row
  residual_ln_kernel<<<Bn * Sn, 256, 0, stream>>>(Ctx, feat, gamma, beta, out);
}
